// MinkUNetBase_68350109548654
// MI455X (gfx1250) — compile-verified, matrix operand order checked by reading, not run on hardware
//
#include <hip/hip_runtime.h>

#define NVOX  200000
#define CCH   64
#define KOFF  27
#define MPAIR 100000
#define EPSV  1e-5f
#define WELEM (KOFF * CCH * CCH)       // 110592 elements per weight tensor

typedef __bf16 bf16_t;
typedef bf16_t v16bf __attribute__((ext_vector_type(16)));
typedef float  v8f   __attribute__((ext_vector_type(8)));

// ---------------------------------------------------------------- zero scratch
__global__ void zero_f4_kernel(float4* __restrict__ p, long n4) {
    long i      = blockIdx.x * (long)blockDim.x + threadIdx.x;
    long stride = (long)gridDim.x * blockDim.x;
    float4 z = {0.f, 0.f, 0.f, 0.f};
    for (; i < n4; i += stride) p[i] = z;
}

// ------------------------------------------- pre-pack weights to B-frag order
// dst layout: [k][tile=ct2*4+nt][lane][e] bf16, where element e of lane
// (h=lane>>4, n=nt*16+(lane&15)) holds w[k][ct+8h+(e&7)+16*(e>=8)][n].
__global__ void prepack_w_kernel(const float* __restrict__ w,
                                 bf16_t* __restrict__ dst) {
    int i = blockIdx.x * blockDim.x + threadIdx.x;
    if (i >= WELEM) return;
    const int k    = i >> 12;
    const int r    = i & 4095;
    const int tile = r >> 9;          // 0..7
    const int lane = (r >> 4) & 31;
    const int e    = r & 15;
    const int ct   = (tile >> 2) * 32;
    const int n    = (tile & 3) * 16 + (lane & 15);
    const int h    = lane >> 4;
    const int kk   = ct + 8 * h + (e & 7) + ((e >= 8) ? 16 : 0);
    dst[i] = (bf16_t)w[(size_t)k * (CCH * CCH) + kk * CCH + n];
}

// --------------------------------------------------------- sparse conv (WMMA)
// One wave = two 16-row tiles of the gathered matrix for offset k.
// A (16x32 bf16): lane (h=lane>>4, m=lane&15); element e holds channel
//   K = ct + 8h + (e&7) + 16*(e>=8)   (CDNA5 ISA 16-bit A layout).
// B: 8 pre-swizzled fragments loaded once per wave into registers.
// C/D (16x16 f32): lane n = lane&15, VGPR v holds row m = 8h + v.
template <bool FUSE_BN_RELU>
__global__ __launch_bounds__(256) void spconv_wmma_kernel(
    const float* __restrict__ src, const bf16_t* __restrict__ wswz,
    const int* __restrict__ in_map, const int* __restrict__ out_map,
    float* __restrict__ dst,
    const float* __restrict__ gamma, const float* __restrict__ beta,
    const float* __restrict__ mean, const float* __restrict__ var)
{
    __shared__ float sArr[CCH];
    __shared__ float bArr[CCH];
    if (FUSE_BN_RELU) {
        if (threadIdx.x < CCH) {
            float s = gamma[threadIdx.x] * rsqrtf(var[threadIdx.x] + EPSV);
            sArr[threadIdx.x] = s;
            bArr[threadIdx.x] = beta[threadIdx.x] - mean[threadIdx.x] * s;
        }
        __syncthreads();
    }

    const int k    = blockIdx.y;
    const int lane = threadIdx.x & 31;
    const int wave = threadIdx.x >> 5;
    const int h    = lane >> 4;
    const int ncol = lane & 15;

    // 8 B fragments: coalesced b128 loads from pre-swizzled bf16 (L2-resident)
    const bf16_t* wk = wswz + (size_t)k * (8 * 32 * 16);
    v16bf bfrag[8];
#pragma unroll
    for (int t = 0; t < 8; ++t)
        bfrag[t] = *(const v16bf*)(wk + ((t * 32 + lane) << 4));

    const int base0 = (blockIdx.x * 8 + wave) * 32;    // two 16-row tiles

#pragma unroll
    for (int halfT = 0; halfT < 2; ++halfT) {
        const int base = base0 + halfT * 16;
        if (base >= MPAIR) break;                      // wave-uniform

        // gather index for this lane's A row; out index for row (lane&15)
        const int gidx = in_map[(size_t)k * MPAIR + base + ncol];
        const int oidx = out_map[(size_t)k * MPAIR + base + ncol];
        const float* row = src + (size_t)gidx * CCH;

        v8f acc[4];
#pragma unroll
        for (int t = 0; t < 4; ++t) acc[t] = (v8f)0.0f;

#pragma unroll
        for (int ct2 = 0; ct2 < 2; ++ct2) {
            const int ct = ct2 * 32;
            // contiguous 32B-aligned channel runs [ct+8h..+7], [ct+16+8h..+7]
            float4 q0 = *(const float4*)(row + ct + 8 * h);
            float4 q1 = *(const float4*)(row + ct + 8 * h + 4);
            float4 q2 = *(const float4*)(row + ct + 16 + 8 * h);
            float4 q3 = *(const float4*)(row + ct + 16 + 8 * h + 4);
            float av[16] = {q0.x, q0.y, q0.z, q0.w, q1.x, q1.y, q1.z, q1.w,
                            q2.x, q2.y, q2.z, q2.w, q3.x, q3.y, q3.z, q3.w};
            v16bf a;
#pragma unroll
            for (int e = 0; e < 16; ++e) {
                float v = av[e];
                if (FUSE_BN_RELU) {
                    const int c = ct + 8 * h + (e & 7) + ((e >= 8) ? 16 : 0);
                    v = fmaxf(v * sArr[c] + bArr[c], 0.0f);
                }
                a[e] = (bf16_t)v;
            }
#pragma unroll
            for (int t = 0; t < 4; ++t) {
                acc[t] = __builtin_amdgcn_wmma_f32_16x16x32_bf16(
                    false, a, false, bfrag[ct2 * 4 + t], (short)0, acc[t],
                    false, false);
            }
        }

        // scatter-add: row m's out index lives in lane m -> wave shuffle
#pragma unroll
        for (int v = 0; v < 8; ++v) {
            const int m   = 8 * h + v;
            const int oid = __shfl(oidx, m, 32);
            float* orow   = dst + (size_t)oid * CCH + ncol;
            unsafeAtomicAdd(orow + 0,  acc[0][v]);
            unsafeAtomicAdd(orow + 16, acc[1][v]);
            unsafeAtomicAdd(orow + 32, acc[2][v]);
            unsafeAtomicAdd(orow + 48, acc[3][v]);
        }
    }
}

// ----------------------------------------------- BN2 + residual + ReLU finale
__global__ void bn_res_relu_kernel(
    const float4* __restrict__ t2, const float4* __restrict__ x,
    const float* __restrict__ gamma, const float* __restrict__ beta,
    const float* __restrict__ mean, const float* __restrict__ var,
    float4* __restrict__ out, int n4)
{
    __shared__ float sArr[CCH];
    __shared__ float bArr[CCH];
    if (threadIdx.x < CCH) {
        float s = gamma[threadIdx.x] * rsqrtf(var[threadIdx.x] + EPSV);
        sArr[threadIdx.x] = s;
        bArr[threadIdx.x] = beta[threadIdx.x] - mean[threadIdx.x] * s;
    }
    __syncthreads();
    int i      = blockIdx.x * blockDim.x + threadIdx.x;
    int stride = gridDim.x * blockDim.x;
    for (; i < n4; i += stride) {
        const int c0 = (i * 4) & (CCH - 1);
        float4 t = t2[i];
        float4 r = x[i];
        float4 o;
        o.x = fmaxf(t.x * sArr[c0 + 0] + bArr[c0 + 0] + r.x, 0.0f);
        o.y = fmaxf(t.y * sArr[c0 + 1] + bArr[c0 + 1] + r.y, 0.0f);
        o.z = fmaxf(t.z * sArr[c0 + 2] + bArr[c0 + 2] + r.z, 0.0f);
        o.w = fmaxf(t.w * sArr[c0 + 3] + bArr[c0 + 3] + r.w, 0.0f);
        out[i] = o;
    }
}

// ---------------------------------------------------------------------- entry
extern "C" void kernel_launch(void* const* d_in, const int* in_sizes, int n_in,
                              void* d_out, int out_size, void* d_ws, size_t ws_size,
                              hipStream_t stream) {
    const float* x      = (const float*)d_in[0];
    const float* w1     = (const float*)d_in[1];
    const float* w2     = (const float*)d_in[2];
    const float* gamma1 = (const float*)d_in[3];
    const float* beta1  = (const float*)d_in[4];
    const float* mean1  = (const float*)d_in[5];
    const float* var1   = (const float*)d_in[6];
    const float* gamma2 = (const float*)d_in[7];
    const float* beta2  = (const float*)d_in[8];
    const float* mean2  = (const float*)d_in[9];
    const float* var2   = (const float*)d_in[10];
    const int*   in_map  = (const int*)d_in[11];
    const int*   out_map = (const int*)d_in[12];

    float* out  = (float*)d_out;
    float* tmp1 = (float*)d_ws;                       // [N, C] conv1 accumulator
    float* tmp2 = tmp1 + (size_t)NVOX * CCH;          // [N, C] conv2 accumulator
    bf16_t* wp1 = (bf16_t*)(tmp2 + (size_t)NVOX * CCH);  // packed w1 (bf16)
    bf16_t* wp2 = wp1 + WELEM;                           // packed w2 (bf16)

    // zero both scatter accumulators (51.2 MB each)
    const long n4z = (long)NVOX * CCH * 2 / 4;
    zero_f4_kernel<<<2048, 256, 0, stream>>>((float4*)tmp1, n4z);

    // pre-pack both weight tensors into bf16 B-fragment layout (221 KB each)
    prepack_w_kernel<<<(WELEM + 255) / 256, 256, 0, stream>>>(w1, wp1);
    prepack_w_kernel<<<(WELEM + 255) / 256, 256, 0, stream>>>(w2, wp2);

    // 6250 16-row tiles per offset, 2 tiles/wave, 8 waves/block
    dim3 grid((MPAIR / 16 + 15) / 16, KOFF);

    spconv_wmma_kernel<false><<<grid, 256, 0, stream>>>(
        x, wp1, in_map, out_map, tmp1,
        nullptr, nullptr, nullptr, nullptr);

    spconv_wmma_kernel<true><<<grid, 256, 0, stream>>>(
        tmp1, wp2, in_map, out_map, tmp2,
        gamma1, beta1, mean1, var1);

    bn_res_relu_kernel<<<2048, 256, 0, stream>>>(
        (const float4*)tmp2, (const float4*)x,
        gamma2, beta2, mean2, var2,
        (float4*)out, (int)((long)NVOX * CCH / 4));
}
